// CGCNNModel_40484361732757
// MI455X (gfx1250) — matrix operand order repaired and checked by reading, not compiled, not run on hardware
//
#include <hip/hip_runtime.h>
#include <hip/hip_bf16.h>

typedef __attribute__((ext_vector_type(8)))  float  v8f;
typedef __attribute__((ext_vector_type(16))) __bf16 v16bf;

#define N_NODES 100000
#define N_EDGES 800000
#define N_GRAPHS 1000
#define F_NODE 92
#define F_EDGE 80
#define H 128
#define Z 336
#define ZP 352          // padded K (11 chunks of 32)
#define NCHUNK 11
#define TILE_E 32       // edges per workgroup
#define BN_EPS 1e-5f

struct H16 { unsigned long long a, b; };           // 16 bytes
struct B32s { H16 lo, hi; };                       // 32 bytes

__device__ inline unsigned short f2bf(float f) {   // RNE f32 -> bf16
    unsigned int u = __float_as_uint(f);
    unsigned int r = u + 0x7FFFu + ((u >> 16) & 1u);
    return (unsigned short)(r >> 16);
}
__device__ inline uint2 pack4(float4 v) {
    uint2 r;
    r.x = (unsigned)f2bf(v.x) | ((unsigned)f2bf(v.y) << 16);
    r.y = (unsigned)f2bf(v.z) | ((unsigned)f2bf(v.w) << 16);
    return r;
}
__device__ inline float sigmoidf_(float x) { return 1.0f / (1.0f + __expf(-x)); }
__device__ inline float softplusf_(float x) { return (x > 20.0f) ? x : log1pf(__expf(x)); }

// ---------------- zero fill ----------------
__global__ void zero_f32(float* p, int n) {
    for (int i = blockIdx.x * blockDim.x + threadIdx.x; i < n; i += gridDim.x * blockDim.x)
        p[i] = 0.0f;
}

// ---------------- weight fragment swizzle ----------------
// dst layout: [(l*2+mat)][chunk c][ntile wv][lane][k]  (16 bf16 = 32B per lane, contiguous)
// fragment element (lane,k) = W[K = 32c + (k<8 ? (lane>>4)*8 + k : 16 + (lane>>4)*8 + k-8)][n = 16wv + (lane&15)], 0 if K>=336
__global__ void prep_wfrag(const float* __restrict__ Wf, const float* __restrict__ Ws,
                           unsigned short* __restrict__ dst) {
    int idx = blockIdx.x * blockDim.x + threadIdx.x;
    if (idx >= 3 * 2 * NCHUNK * 8 * 32) return;
    int lane = idx & 31;
    int r = idx >> 5;
    int wv = r & 7;  r >>= 3;
    int c = r % NCHUNK; r /= NCHUNK;
    int mat = r & 1;
    int l = r >> 1;
    const float* W = (mat ? Ws : Wf) + (size_t)l * Z * H;
    int n = wv * 16 + (lane & 15);
    unsigned short* out = dst + (size_t)idx * 16;
    for (int k = 0; k < 16; ++k) {
        int K = 32 * c + ((k < 8) ? (lane >> 4) * 8 + k : 16 + (lane >> 4) * 8 + (k - 8));
        float v = (K < Z) ? W[(size_t)K * H + n] : 0.0f;
        out[k] = f2bf(v);
    }
}

// ---------------- node embedding: h = x @ embW + b ----------------
__global__ void embed_kernel(const float* __restrict__ x, const float* __restrict__ W,
                             const float* __restrict__ b, float* __restrict__ h) {
    __shared__ float xs[F_NODE];
    int n = blockIdx.x, c = threadIdx.x;
    if (c < F_NODE) xs[c] = x[(size_t)n * F_NODE + c];
    __syncthreads();
    float acc = b[c];
    #pragma unroll 4
    for (int k = 0; k < F_NODE; ++k) acc += xs[k] * W[(size_t)k * H + c];
    h[(size_t)n * H + c] = acc;
}

// ---------------- edge conv: WMMA bf16 gate/core GEMM + sigmoid*softplus scatter ----------------
__global__ void __launch_bounds__(256, 2)
edge_conv(const float* __restrict__ h, const float* __restrict__ ea,
          const int* __restrict__ ei,
          const unsigned short* __restrict__ wfF, const unsigned short* __restrict__ wfS,
          const float* __restrict__ bF, const float* __restrict__ bS,
          float* __restrict__ agg) {
    __shared__ __align__(16) unsigned short zb[TILE_E * 360]; // 360 ushort = 720B row stride
    __shared__ int dstL[TILE_E];

    const int tid  = threadIdx.x;
    const int lane = tid & 31;
    const int wv   = tid >> 5;                  // wave id 0..7 == N-tile
    const long long tbase = (long long)blockIdx.x * TILE_E;

    // ---- gather 4 edges per wave into LDS (bf16) ----
    for (int i = 0; i < 4; ++i) {
        int el = wv * 4 + i;
        long long e = tbase + el;
        int s = ei[e];
        int d = ei[N_EDGES + e];
        if (lane == 0) dstL[el] = d;
        unsigned short* row = &zb[el * 360];
        float4 hd = *((const float4*)(h + (size_t)d * H) + lane);
        *(uint2*)(row + lane * 4) = pack4(hd);                       // cols 0..127
        float4 hs = *((const float4*)(h + (size_t)s * H) + lane);
        *(uint2*)(row + 128 + lane * 4) = pack4(hs);                 // cols 128..255
        if (lane < 20) {
            float4 ev = *((const float4*)(ea + (size_t)e * F_EDGE) + lane);
            *(uint2*)(row + 256 + lane * 4) = pack4(ev);             // cols 256..335
        } else if (lane < 24) {
            uint2 zz; zz.x = 0u; zz.y = 0u;
            *(uint2*)(row + 336 + (lane - 20) * 4) = zz;             // pad cols 336..351
        }
    }
    __syncthreads();

    // ---- WMMA accumulate over 11 K-chunks ----
    v8f g0 = {}, g1 = {}, s0 = {}, s1 = {};
    const int m  = lane & 15;
    const int hh = lane >> 4;
    const int u1base = hh * 8;                  // ushort offset of first 8-K run
    #pragma unroll
    for (int c = 0; c < NCHUNK; ++c) {
        int u1 = c * 32 + u1base;
        const unsigned short* r0 = &zb[m * 360 + u1];
        const unsigned short* r1 = &zb[(16 + m) * 360 + u1];
        B32s a0s, a1s;
        a0s.lo = *(const H16*)r0;        a0s.hi = *(const H16*)(r0 + 16);
        a1s.lo = *(const H16*)r1;        a1s.hi = *(const H16*)(r1 + 16);
        v16bf a0 = __builtin_bit_cast(v16bf, a0s);
        v16bf a1 = __builtin_bit_cast(v16bf, a1s);
        const unsigned short* wb = wfF + ((size_t)(c * 8 + wv) * 32 + lane) * 16;
        v16bf bf = __builtin_bit_cast(v16bf, *(const B32s*)wb);
        const unsigned short* ws = wfS + ((size_t)(c * 8 + wv) * 32 + lane) * 16;
        v16bf bs = __builtin_bit_cast(v16bf, *(const B32s*)ws);
        g0 = __builtin_amdgcn_wmma_f32_16x16x32_bf16(false, a0, false, bf, (short)0, g0, false, false);
        g1 = __builtin_amdgcn_wmma_f32_16x16x32_bf16(false, a1, false, bf, (short)0, g1, false, false);
        s0 = __builtin_amdgcn_wmma_f32_16x16x32_bf16(false, a0, false, bs, (short)0, s0, false, false);
        s1 = __builtin_amdgcn_wmma_f32_16x16x32_bf16(false, a1, false, bs, (short)0, s1, false, false);
    }

    // ---- epilogue: bias, sigmoid*softplus, scatter-add by dst ----
    const int n = wv * 16 + (lane & 15);
    const float bfv = bF[n];
    const float bsv = bS[n];
    #pragma unroll
    for (int r = 0; r < 8; ++r) {
        int row0 = hh * 8 + r;          // edge row within tile, M-tile 0
        float v0 = sigmoidf_(g0[r] + bfv) * softplusf_(s0[r] + bsv);
        unsafeAtomicAdd(&agg[(size_t)dstL[row0] * H + n], v0);
        int row1 = 16 + row0;           // M-tile 1
        float v1 = sigmoidf_(g1[r] + bfv) * softplusf_(s1[r] + bsv);
        unsafeAtomicAdd(&agg[(size_t)dstL[row1] * H + n], v1);
    }
}

// ---------------- batch-norm stats (sum, sumsq per column) ----------------
__global__ void bn_stats(const float* __restrict__ agg, float* __restrict__ stats) {
    int col = threadIdx.x & 127;
    int sub = threadIdx.x >> 7;
    float s = 0.0f, ss = 0.0f;
    for (int r = blockIdx.x * 2 + sub; r < N_NODES; r += gridDim.x * 2) {
        float v = agg[(size_t)r * H + col];
        s += v; ss += v * v;
    }
    unsafeAtomicAdd(&stats[col], s);
    unsafeAtomicAdd(&stats[H + col], ss);
}

// ---------------- BN apply + residual; re-zero agg for next layer ----------------
__global__ void bn_apply(float* __restrict__ h, float* __restrict__ agg,
                         const float* __restrict__ stats,
                         const float* __restrict__ gamma, const float* __restrict__ beta) {
    const float invN = 1.0f / (float)N_NODES;
    for (int i = blockIdx.x * blockDim.x + threadIdx.x; i < N_NODES * H;
         i += gridDim.x * blockDim.x) {
        int col = i & 127;
        float mean = stats[col] * invN;
        float var  = stats[H + col] * invN - mean * mean;
        float a = agg[i];
        h[i] += gamma[col] * (a - mean) * rsqrtf(var + BN_EPS) + beta[col];
        agg[i] = 0.0f;
    }
}

// ---------------- global mean pool (scatter) ----------------
__global__ void pool_kernel(const float* __restrict__ h, const int* __restrict__ batch,
                            float* __restrict__ gsum, float* __restrict__ cnt) {
    for (int i = blockIdx.x * blockDim.x + threadIdx.x; i < N_NODES * H;
         i += gridDim.x * blockDim.x) {
        int n = i >> 7, c = i & 127;
        int g = batch[n];
        unsafeAtomicAdd(&gsum[(size_t)g * H + c], h[i]);
        if (c == 0) unsafeAtomicAdd(&cnt[g], 1.0f);
    }
}

// ---------------- per-graph head: mean, fc+softplus, out ----------------
__global__ void head_kernel(const float* __restrict__ gsum, const float* __restrict__ cnt,
                            const float* __restrict__ fcW, const float* __restrict__ fcb,
                            const float* __restrict__ outW, const float* __restrict__ outb,
                            float* __restrict__ out) {
    __shared__ float gv[H];
    __shared__ float sv[H];
    int g = blockIdx.x, c = threadIdx.x;
    float cg = cnt[g];
    gv[c] = gsum[(size_t)g * H + c] / fmaxf(cg, 1.0f);
    __syncthreads();
    float acc = fcb[c];
    #pragma unroll 4
    for (int k = 0; k < H; ++k) acc += gv[k] * fcW[(size_t)k * H + c];
    sv[c] = softplusf_(acc);
    __syncthreads();
    gv[c] = sv[c] * outW[c];
    __syncthreads();
    for (int s = 64; s > 0; s >>= 1) {
        if (c < s) gv[c] += gv[c + s];
        __syncthreads();
    }
    if (c == 0) out[g] = gv[0] + outb[0];
}

extern "C" void kernel_launch(void* const* d_in, const int* in_sizes, int n_in,
                              void* d_out, int out_size, void* d_ws, size_t ws_size,
                              hipStream_t stream) {
    const float* x     = (const float*)d_in[0];
    const float* ea    = (const float*)d_in[1];
    const float* embW  = (const float*)d_in[2];
    const float* embB  = (const float*)d_in[3];
    const float* linfW = (const float*)d_in[4];
    const float* linfB = (const float*)d_in[5];
    const float* linsW = (const float*)d_in[6];
    const float* linsB = (const float*)d_in[7];
    const float* bng   = (const float*)d_in[8];
    const float* bnb   = (const float*)d_in[9];
    const float* fcW   = (const float*)d_in[10];
    const float* fcb   = (const float*)d_in[11];
    const float* outW  = (const float*)d_in[12];
    const float* outb  = (const float*)d_in[13];
    const int*   ei    = (const int*)d_in[14];
    const int*   batch = (const int*)d_in[15];
    float* out = (float*)d_out;

    char* ws = (char*)d_ws;
    float* h     = (float*)(ws);                              // 51,200,000 B
    float* agg   = (float*)(ws + 51200000);                   // 51,200,000 B
    float* gsum  = (float*)(ws + 102400000);                  // 512,000 B
    float* cnt   = (float*)(ws + 102912000);                  // 4,000 B (pad 4096)
    float* stats = (float*)(ws + 102916096);                  // 3*256*4 B (pad 4096)
    unsigned short* wfrag = (unsigned short*)(ws + 102920192);// 540,672 B

    // init scratch (agg re-zeroed by bn_apply each layer; others written once per launch)
    zero_f32<<<4096, 256, 0, stream>>>(agg, N_NODES * H);
    zero_f32<<<512, 256, 0, stream>>>(gsum, N_GRAPHS * H);
    zero_f32<<<8, 256, 0, stream>>>(cnt, N_GRAPHS);
    zero_f32<<<4, 256, 0, stream>>>(stats, 3 * 2 * H);

    // bf16 weight fragments for all 3 layers x {gate,core}
    prep_wfrag<<<66, 256, 0, stream>>>(linfW, linsW, wfrag);

    // node embedding
    embed_kernel<<<N_NODES, H, 0, stream>>>(x, embW, embB, h);

    const size_t layerFrag = (size_t)NCHUNK * 8 * 32 * 16;    // ushorts per (layer,mat)
    for (int l = 0; l < 3; ++l) {
        const unsigned short* wfF = wfrag + (size_t)(l * 2 + 0) * layerFrag;
        const unsigned short* wfS = wfrag + (size_t)(l * 2 + 1) * layerFrag;
        edge_conv<<<N_EDGES / TILE_E, 256, 0, stream>>>(
            h, ea, ei, wfF, wfS, linfB + l * H, linsB + l * H, agg);
        bn_stats<<<200, 256, 0, stream>>>(agg, stats + l * 2 * H);
        bn_apply<<<50000, 256, 0, stream>>>(h, agg, stats + l * 2 * H,
                                            bng + l * H, bnb + l * H);
    }

    pool_kernel<<<50000, 256, 0, stream>>>(h, batch, gsum, cnt);
    head_kernel<<<N_GRAPHS, H, 0, stream>>>(gsum, cnt, fcW, fcb, outW, outb, out);
}
